// EpiSIGNetV2_NoSIG_32830730010777
// MI455X (gfx1250) — matrix operand order repaired and checked by reading, not compiled, not run on hardware
//
#include <hip/hip_runtime.h>
#include <hip/hip_bf16.h>
#include <hip/hip_fp16.h>

typedef __attribute__((ext_vector_type(16))) _Float16 v16h;
typedef __attribute__((ext_vector_type(8)))  float    v8f;
typedef _Float16 half_t;

#define DEV static __device__ __forceinline__

// ---------------------------------------------------------------------------
// WMMA fragment helpers (CDNA5 v_wmma_f32_16x16x32_f16, wave32)
// A (16x32 f16): lanes 0-15 -> row M=lane, K {k0..k0+7, k0+16..k0+23}
//                lanes 16-31 -> row M=lane-16, K {k0+8..k0+15, k0+24..k0+31}
// B (32x16 f16): lanes 0-15 -> col N=lane, K {k0..k0+7, k0+16..k0+23}
//                lanes 16-31 -> col N=lane-16, K {k0+8..k0+15, k0+24..k0+31}
// C (16x16 f32): VGPR r -> M = r + 8*(lane>=16), N = lane&15
// ---------------------------------------------------------------------------
DEV v16h load_a_f32(const float* __restrict__ A, int row, int ld, int k0, int lane) {
  const int hs = (lane >> 4) & 1;
  const float* p = A + (size_t)row * ld + k0 + hs * 8;
  v16h a;
#pragma unroll
  for (int i = 0; i < 8; ++i) { a[i] = (_Float16)p[i]; a[8 + i] = (_Float16)p[16 + i]; }
  return a;
}

DEV v16h load_a_f16(const half_t* __restrict__ A, int row, int ld, int k0, int lane) {
  const int hs = (lane >> 4) & 1;
  const half_t* p = A + (size_t)row * ld + k0 + hs * 8;
  v16h a;
#pragma unroll
  for (int i = 0; i < 8; ++i) { a[i] = p[i]; a[8 + i] = p[16 + i]; }
  return a;
}

DEV v16h load_bt_f16(const half_t* __restrict__ BT, int col, int ld, int k0, int lane) {
  const int hs = (lane >> 4) & 1;
  const half_t* p = BT + (size_t)col * ld + k0 + hs * 8;
  v16h b;
#pragma unroll
  for (int i = 0; i < 8; ++i) { b[i] = p[i]; b[8 + i] = p[16 + i]; }
  return b;
}

DEV v8f wmma_acc(v16h a, v16h b, v8f c) {
  return __builtin_amdgcn_wmma_f32_16x16x32_f16(false, a, false, b, (short)0, c, false, false);
}

DEV float wave_allreduce(float v) {
#pragma unroll
  for (int o = 16; o > 0; o >>= 1) v += __shfl_xor(v, o, 32);
  return v;
}

// ---------------------------------------------------------------------------
// Weight transpose + fp32 -> fp16  (W[k][n] -> WT[n][k]), 256x256
// ---------------------------------------------------------------------------
__global__ __launch_bounds__(256) void trans_kernel(const float* __restrict__ W,
                                                    half_t* __restrict__ WT) {
  int idx = blockIdx.x * 256 + threadIdx.x;   // 65536
  int n = idx >> 8, k = idx & 255;
  WT[n * 256 + k] = (half_t)W[k * 256 + n];
}

// ---------------------------------------------------------------------------
// Stage 1: DS-conv stack + proj_low/high + encoder LayerNorm
// one block of 128 threads per (b, n) sequence of length T=96
// ---------------------------------------------------------------------------
struct ConvParams {
  const float* x;
  const float* fc_dww; const float* fc_dwb; const float* fc_bn1g; const float* fc_bn1b;
  const float* fc_pww; const float* fc_pwb; const float* fc_bn2g; const float* fc_bn2b;
  const float* sc_dww[3]; const float* sc_dwb[3]; const float* sc_bn1g[3]; const float* sc_bn1b[3];
  const float* sc_pww[3]; const float* sc_pwb[3]; const float* sc_bn2g[3]; const float* sc_bn2b[3];
  const float* scale_w;
  const float* plw; const float* plb; const float* phw; const float* phb;
  const float* lng; const float* lnb;
  float* h32;
};

__global__ __launch_bounds__(128) void conv_proj_kernel(ConvParams P) {
  __shared__ float xs[96];
  __shared__ float a0[96];
  __shared__ float feat[16][96];
  __shared__ float tmp[16][96];
  __shared__ float fus[16][96];
  __shared__ float h1[8];
  __shared__ float red[256];
  const int row = blockIdx.x;              // b*2048 + n
  const int b = row >> 11, n = row & 2047;
  const int tid = threadIdx.x;
  const float bns = rsqrtf(1.0f + 1e-5f);

  // softmax(scale_weights) — uniform across threads, 3 values
  float s0 = P.scale_w[0], s1 = P.scale_w[1], s2 = P.scale_w[2];
  float mx = fmaxf(s0, fmaxf(s1, s2));
  float e0 = __expf(s0 - mx), e1 = __expf(s1 - mx), e2 = __expf(s2 - mx);
  float einv = 1.f / (e0 + e1 + e2);
  float wgt[3] = {e0 * einv, e1 * einv, e2 * einv};

  for (int t = tid; t < 96; t += 128) xs[t] = P.x[((size_t)(b * 96 + t)) * 2048 + n];
  __syncthreads();

  // feature_conv: depthwise (1ch, K=3, pad 1) -> bn -> relu
  for (int t = tid; t < 96; t += 128) {
    float acc = P.fc_dwb[0];
#pragma unroll
    for (int k = 0; k < 3; ++k) {
      int tt = t + k - 1;
      if (tt >= 0 && tt < 96) acc += xs[tt] * P.fc_dww[k];
    }
    acc = acc * (P.fc_bn1g[0] * bns) + P.fc_bn1b[0];
    a0[t] = fmaxf(acc, 0.f);
  }
  __syncthreads();
  // pointwise 1 -> 16 -> bn -> relu
  for (int i = tid; i < 1536; i += 128) {
    int c = i / 96, t = i - c * 96;
    float v = a0[t] * P.fc_pww[c] + P.fc_pwb[c];
    v = v * (P.fc_bn2g[c] * bns) + P.fc_bn2b[c];
    feat[c][t] = fmaxf(v, 0.f);
    fus[c][t] = 0.f;
  }
  __syncthreads();

  // three dilated DS convs, weighted-summed
  for (int s = 0; s < 3; ++s) {
    const int dil = 1 << s;
    for (int i = tid; i < 1536; i += 128) {
      int c = i / 96, t = i - c * 96;
      float acc = P.sc_dwb[s][c];
#pragma unroll
      for (int k = 0; k < 3; ++k) {
        int tt = t + (k - 1) * dil;
        if (tt >= 0 && tt < 96) acc += feat[c][tt] * P.sc_dww[s][c * 3 + k];
      }
      acc = acc * (P.sc_bn1g[s][c] * bns) + P.sc_bn1b[s][c];
      tmp[c][t] = fmaxf(acc, 0.f);
    }
    __syncthreads();
    for (int i = tid; i < 1536; i += 128) {
      int c = i / 96, t = i - c * 96;
      float acc = P.sc_pwb[s][c];
#pragma unroll
      for (int ci = 0; ci < 16; ++ci) acc += tmp[ci][t] * P.sc_pww[s][c * 16 + ci];
      acc = acc * (P.sc_bn2g[s][c] * bns) + P.sc_bn2b[s][c];
      fus[c][t] += wgt[s] * fmaxf(acc, 0.f);
    }
    __syncthreads();
  }

  // proj_low: (1536 -> 8), block reduction via LDS float atomics (ds_add_f32)
  if (tid < 8) h1[tid] = P.plb[tid];
  float pacc[8] = {0, 0, 0, 0, 0, 0, 0, 0};
  __syncthreads();
  for (int i = tid; i < 1536; i += 128) {
    int c = i / 96, t = i - c * 96;
    float f = fus[c][t];
#pragma unroll
    for (int j = 0; j < 8; ++j) pacc[j] += f * P.plw[i * 8 + j];
  }
#pragma unroll
  for (int j = 0; j < 8; ++j) atomicAdd(&h1[j], pacc[j]);
  __syncthreads();

  // proj_high (8 -> 256) + LayerNorm
  float* h2 = (float*)tmp;  // reuse LDS
  float ss = 0.f, sq = 0.f;
  for (int d = tid; d < 256; d += 128) {
    float acc = P.phb[d];
#pragma unroll
    for (int j = 0; j < 8; ++j) acc += h1[j] * P.phw[j * 256 + d];
    h2[d] = acc; ss += acc; sq += acc * acc;
  }
  red[tid] = ss; red[128 + tid] = sq;
  __syncthreads();
  for (int o = 64; o > 0; o >>= 1) {
    if (tid < o) { red[tid] += red[tid + o]; red[128 + tid] += red[128 + tid + o]; }
    __syncthreads();
  }
  float mean = red[0] * (1.f / 256.f);
  float var = red[128] * (1.f / 256.f) - mean * mean;
  float inv = rsqrtf(var + 1e-5f);
  for (int d = tid; d < 256; d += 128)
    P.h32[(size_t)row * 256 + d] = (h2[d] - mean) * inv * P.lng[d] + P.lnb[d];
}

// ---------------------------------------------------------------------------
// Stage 2: q/k/v = h @ {Wq,Wk,Wv}
// One wave per 16-row strip; full-K A strip held in registers (8 x v16h),
// swept against all 16 column tiles of all three weight matrices.
// epilogue: elu+1 on q/k; write kT,vT in f16 (per-batch (d,n)/(e,n) layout)
// ---------------------------------------------------------------------------
__global__ __launch_bounds__(256) void qkv_kernel(
    const float* __restrict__ h32,
    const half_t* __restrict__ WqT, const half_t* __restrict__ WkT,
    const half_t* __restrict__ WvT,
    float* __restrict__ q32, half_t* __restrict__ kT16, half_t* __restrict__ vT16) {
  const int lane = threadIdx.x & 31;
  const int wave = threadIdx.x >> 5;
  const int strip = blockIdx.x * 8 + wave;      // 2048 strips of 16 rows
  const int row0 = strip * 16;
  const int ar = row0 + (lane & 15);
  const int hs = (lane >> 4) & 1;

  // warm L2 with the weight tables
  __builtin_prefetch(WqT + (size_t)lane * 512, 0, 1);
  __builtin_prefetch(WkT + (size_t)lane * 512, 0, 1);
  __builtin_prefetch(WvT + (size_t)lane * 512, 0, 1);

  v16h afr[8];
#pragma unroll
  for (int kk = 0; kk < 8; ++kk) afr[kk] = load_a_f32(h32, ar, 256, kk * 32, lane);

  const int mb = row0 >> 11;                    // batch (strip fits in one batch)
  for (int tn = 0; tn < 16; ++tn) {
    const int bc = tn * 16 + (lane & 15);
    const int nc = tn * 16 + (lane & 15);
    // ---- q ----
    {
      v8f c = {};
#pragma unroll
      for (int kk = 0; kk < 8; ++kk)
        c = wmma_acc(afr[kk], load_bt_f16(WqT, bc, 256, kk * 32, lane), c);
#pragma unroll
      for (int r = 0; r < 8; ++r) {
        int m = row0 + r + 8 * hs;
        float qv = c[r]; qv = qv > 0.f ? qv + 1.f : __expf(qv);   // elu(x)+1
        q32[(size_t)m * 256 + nc] = qv;
      }
    }
    // ---- k ----
    {
      v8f c = {};
#pragma unroll
      for (int kk = 0; kk < 8; ++kk)
        c = wmma_acc(afr[kk], load_bt_f16(WkT, bc, 256, kk * 32, lane), c);
#pragma unroll
      for (int r = 0; r < 8; ++r) {
        int m = row0 + r + 8 * hs;
        int nn = m & 2047;
        float kv = c[r]; kv = kv > 0.f ? kv + 1.f : __expf(kv);
        kT16[((size_t)(mb * 256 + nc)) * 2048 + nn] = (half_t)kv;
      }
    }
    // ---- v ----
    {
      v8f c = {};
#pragma unroll
      for (int kk = 0; kk < 8; ++kk)
        c = wmma_acc(afr[kk], load_bt_f16(WvT, bc, 256, kk * 32, lane), c);
#pragma unroll
      for (int r = 0; r < 8; ++r) {
        int m = row0 + r + 8 * hs;
        int nn = m & 2047;
        vT16[((size_t)(mb * 256 + nc)) * 2048 + nn] = (half_t)c[r];
      }
    }
  }
}

// ksum[b][d] = sum_n k[b][n][d]   (reads kT16 row-contiguous)
__global__ __launch_bounds__(256) void ksum_kernel(const half_t* __restrict__ kT16,
                                                   float* __restrict__ ksum) {
  __shared__ float red[256];
  const half_t* p = kT16 + (size_t)blockIdx.x * 2048;   // blockIdx = b*256 + d
  float s = 0.f;
  for (int i = threadIdx.x; i < 2048; i += 256) s += (float)p[i];
  red[threadIdx.x] = s;
  __syncthreads();
  for (int o = 128; o > 0; o >>= 1) {
    if (threadIdx.x < o) red[threadIdx.x] += red[threadIdx.x + o];
    __syncthreads();
  }
  if (threadIdx.x == 0) ksum[blockIdx.x] = red[0];
}

// z[m] = q[m,:] . ksum[b,:]
__global__ __launch_bounds__(256) void z_kernel(const float* __restrict__ q32,
                                                const float* __restrict__ ksum,
                                                float* __restrict__ z) {
  const int lane = threadIdx.x & 31, wave = threadIdx.x >> 5;
  const int row = blockIdx.x * 8 + wave;
  const int b = row >> 11;
  const float* qp = q32 + (size_t)row * 256;
  const float* kp = ksum + b * 256;
  float s = 0.f;
#pragma unroll
  for (int j = 0; j < 8; ++j) { int d = lane + 32 * j; s += qp[d] * kp[d]; }
  s = wave_allreduce(s);
  if (lane == 0) z[row] = s;
}

// kv[b][d][e] = sum_n kT[d][n] * vT[e][n]   (per-batch 256x256, K=2048)
// 4x4 block of 16x16 tiles per wave: 16 accumulators, 4x A + 4x B fragment
// reuse per K step -> 4x less operand traffic on both sides.
__global__ __launch_bounds__(256) void kv_kernel(const half_t* __restrict__ kT16,
                                                 const half_t* __restrict__ vT16,
                                                 half_t* __restrict__ kvT16) {
  const int lane = threadIdx.x & 31;
  const int wave = threadIdx.x >> 5;
  const int g = blockIdx.x * 8 + wave;     // 256 groups: 16 batches x 4 x 4
  const int b = g >> 4;
  const int gd = (g >> 2) & 3, ge = g & 3;
  const half_t* A  = kT16 + (size_t)b * 256 * 2048;
  const half_t* Bt = vT16 + (size_t)b * 256 * 2048;
  const int l15 = lane & 15;
  v8f c[4][4];
#pragma unroll
  for (int i = 0; i < 4; ++i)
#pragma unroll
    for (int j = 0; j < 4; ++j) c[i][j] = (v8f){};

  for (int k0 = 0; k0 < 2048; k0 += 32) {
    v16h a[4];
#pragma unroll
    for (int i = 0; i < 4; ++i)
      a[i] = load_a_f16(A, (gd * 4 + i) * 16 + l15, 2048, k0, lane);
#pragma unroll
    for (int j = 0; j < 4; ++j) {
      v16h bb = load_bt_f16(Bt, (ge * 4 + j) * 16 + l15, 2048, k0, lane);
#pragma unroll
      for (int i = 0; i < 4; ++i) c[i][j] = wmma_acc(a[i], bb, c[i][j]);
    }
  }
  const int hs = (lane >> 4) & 1;
#pragma unroll
  for (int i = 0; i < 4; ++i)
#pragma unroll
    for (int j = 0; j < 4; ++j) {
      const int e = (ge * 4 + j) * 16 + l15;
#pragma unroll
      for (int r = 0; r < 8; ++r) {
        int d = (gd * 4 + i) * 16 + r + 8 * hs;
        kvT16[((size_t)(b * 256 + e)) * 256 + d] = (half_t)c[i][j][r];  // (e,d)
      }
    }
}

// attn[m][e] = (q[m,:] @ kv[:, e]) / (z[m] + 1e-6)
// full-K q strip in registers, swept over 16 column tiles of kv
__global__ __launch_bounds__(256) void attn_kernel(const float* __restrict__ q32,
                                                   const half_t* __restrict__ kvT16,
                                                   const float* __restrict__ z,
                                                   float* __restrict__ attn32) {
  const int lane = threadIdx.x & 31;
  const int wave = threadIdx.x >> 5;
  const int strip = blockIdx.x * 8 + wave;
  const int row0 = strip * 16;
  const int b = row0 >> 11;
  const half_t* Bt = kvT16 + (size_t)b * 256 * 256;
  const int ar = row0 + (lane & 15);
  const int hs = (lane >> 4) & 1;
  __builtin_prefetch(Bt + (size_t)lane * 2048, 0, 1);

  v16h afr[8];
#pragma unroll
  for (int kk = 0; kk < 8; ++kk) afr[kk] = load_a_f32(q32, ar, 256, kk * 32, lane);

  float zr[8];
#pragma unroll
  for (int r = 0; r < 8; ++r) zr[r] = z[row0 + r + 8 * hs] + 1e-6f;

  for (int tn = 0; tn < 16; ++tn) {
    const int bc = tn * 16 + (lane & 15);
    v8f c = {};
#pragma unroll
    for (int kk = 0; kk < 8; ++kk)
      c = wmma_acc(afr[kk], load_bt_f16(Bt, bc, 256, kk * 32, lane), c);
#pragma unroll
    for (int r = 0; r < 8; ++r) {
      int m = row0 + r + 8 * hs;
      attn32[(size_t)m * 256 + bc] = c[r] / zr[r];
    }
  }
}

// s[m][n] = h[m][n] + attn[m,:] @ Wo[:, n]
__global__ __launch_bounds__(256) void attnwo_kernel(const float* __restrict__ attn32,
                                                     const half_t* __restrict__ WoT,
                                                     const float* __restrict__ h32,
                                                     float* __restrict__ s32) {
  const int lane = threadIdx.x & 31;
  const int wave = threadIdx.x >> 5;
  const int strip = blockIdx.x * 8 + wave;
  const int row0 = strip * 16;
  const int ar = row0 + (lane & 15);
  const int hs = (lane >> 4) & 1;
  __builtin_prefetch(WoT + (size_t)lane * 512, 0, 1);

  v16h afr[8];
#pragma unroll
  for (int kk = 0; kk < 8; ++kk) afr[kk] = load_a_f32(attn32, ar, 256, kk * 32, lane);

  for (int tn = 0; tn < 16; ++tn) {
    const int bc = tn * 16 + (lane & 15);
    v8f c = {};
#pragma unroll
    for (int kk = 0; kk < 8; ++kk)
      c = wmma_acc(afr[kk], load_bt_f16(WoT, bc, 256, kk * 32, lane), c);
#pragma unroll
    for (int r = 0; r < 8; ++r) {
      int m = row0 + r + 8 * hs;
      s32[(size_t)m * 256 + bc] = h32[(size_t)m * 256 + bc] + c[r];
    }
  }
}

// row LayerNorm over D=256 (one wave per row, butterfly reductions)
__global__ __launch_bounds__(256) void ln_kernel(const float* __restrict__ s32,
                                                 const float* __restrict__ g,
                                                 const float* __restrict__ bb,
                                                 float* __restrict__ out) {
  const int lane = threadIdx.x & 31, wave = threadIdx.x >> 5;
  const int row = blockIdx.x * 8 + wave;
  const float* p = s32 + (size_t)row * 256;
  float v[8]; float s = 0.f, sq = 0.f;
#pragma unroll
  for (int j = 0; j < 8; ++j) { float t = p[lane + 32 * j]; v[j] = t; s += t; sq += t * t; }
  s = wave_allreduce(s); sq = wave_allreduce(sq);
  float mean = s * (1.f / 256.f);
  float var = sq * (1.f / 256.f) - mean * mean;
  float inv = rsqrtf(var + 1e-5f);
  float* q = out + (size_t)row * 256;
#pragma unroll
  for (int j = 0; j < 8; ++j) { int d = lane + 32 * j; q[d] = (v[j] - mean) * inv * g[d] + bb[d]; }
}

// ---------------------------------------------------------------------------
// Stage 3: 4 MLP heads + fusion + gated decay + highway, one row per wave
// ---------------------------------------------------------------------------
struct HeadParams {
  const float* h3; const float* x;
  const float* sw1; const float* sb1; const float* sw2; const float* sb2;
  const float* mw1; const float* mb1; const float* mw2; const float* mb2;
  const float* lw1; const float* lb1; const float* lw2; const float* lb2;
  const float* rw1; const float* rb1; const float* rw2; const float* rb2;
  const float* fw; const float* fb; const float* log_decay;
  const float* hww; const float* hwb; const float* hwr;
  float* out;
};

__global__ __launch_bounds__(256) void heads_kernel(HeadParams P) {
  const int lane = threadIdx.x & 31, wave = threadIdx.x >> 5;
  const int row = blockIdx.x * 8 + wave;
  const int b = row >> 11, n = row & 2047;
  const float* hp = P.h3 + (size_t)row * 256;
  float hv[8];
#pragma unroll
  for (int j = 0; j < 8; ++j) hv[j] = hp[lane + 32 * j];

  const float* w1s[4] = {P.sw1, P.mw1, P.lw1, P.rw1};
  const float* b1s[4] = {P.sb1, P.mb1, P.lb1, P.rb1};
  float hid[4][8];
  for (int hd = 0; hd < 4; ++hd) {
    const float* w1 = w1s[hd];
#pragma unroll
    for (int j = 0; j < 8; ++j) {
      float part = 0.f;
#pragma unroll
      for (int jj = 0; jj < 8; ++jj) {
        int d = lane + 32 * jj;
        part += hv[jj] * w1[d * 8 + j];
      }
      part = wave_allreduce(part);           // broadcast to all lanes
      hid[hd][j] = fmaxf(part + b1s[hd][j], 0.f);
    }
  }
  if (lane < 12) {
    const int hh = lane;
    float sp = 0.f, mp = 0.f;
    if (hh < 3) { sp = P.sb2[hh]; for (int j = 0; j < 8; ++j) sp += hid[0][j] * P.sw2[j * 3 + hh]; }
    if (hh < 7) { mp = P.mb2[hh]; for (int j = 0; j < 8; ++j) mp += hid[1][j] * P.mw2[j * 7 + hh]; }
    float lp = P.lb2[hh]; for (int j = 0; j < 8; ++j) lp += hid[2][j] * P.lw2[j * 12 + hh];
    float rp = P.rb2[hh]; for (int j = 0; j < 8; ++j) rp += hid[3][j] * P.rw2[j * 12 + hh];
    float raw = P.fw[0] * sp + P.fw[1] * mp + P.fw[2] * lp + P.fb[0];
    float gate = 1.f / (1.f + __expf(-rp));
    float decay = __expf(-__expf(P.log_decay[0]) * (float)(hh + 1));
    float x_last = P.x[((size_t)(b * 96 + 95)) * 2048 + n];
    float pred = gate * raw + (1.f - gate) * x_last * decay;
    float hw = P.hwb[hh];
#pragma unroll
    for (int t = 0; t < 4; ++t)
      hw += P.x[((size_t)(b * 96 + 92 + t)) * 2048 + n] * P.hww[t * 12 + hh];
    float ratio = 1.f / (1.f + __expf(-P.hwr[0]));
    pred = ratio * pred + (1.f - ratio) * hw;
    P.out[((size_t)(b * 12 + hh)) * 2048 + n] = pred;
  }
}

// ---------------------------------------------------------------------------
extern "C" void kernel_launch(void* const* d_in, const int* in_sizes, int n_in,
                              void* d_out, int out_size, void* d_ws, size_t ws_size,
                              hipStream_t stream) {
  (void)in_sizes; (void)n_in; (void)out_size; (void)ws_size;
  const float* x = (const float*)d_in[0];
  int idx = 1;
  const float* fc[8];
  for (int i = 0; i < 8; ++i) fc[i] = (const float*)d_in[idx++];
  const float* sc[3][8];
  for (int s = 0; s < 3; ++s)
    for (int i = 0; i < 8; ++i) sc[s][i] = (const float*)d_in[idx++];
  const float* scale_w = (const float*)d_in[idx++];
  const float* plw = (const float*)d_in[idx++];
  const float* plb = (const float*)d_in[idx++];
  const float* phw = (const float*)d_in[idx++];
  const float* phb = (const float*)d_in[idx++];
  const float* lng = (const float*)d_in[idx++];
  const float* lnb = (const float*)d_in[idx++];
  const float* Wq = (const float*)d_in[idx++];
  const float* Wk = (const float*)d_in[idx++];
  const float* Wv = (const float*)d_in[idx++];
  const float* Wo = (const float*)d_in[idx++];
  const float* aln_g = (const float*)d_in[idx++];
  const float* aln_b = (const float*)d_in[idx++];
  const float* sw1 = (const float*)d_in[idx++]; const float* sb1 = (const float*)d_in[idx++];
  const float* sw2 = (const float*)d_in[idx++]; const float* sb2 = (const float*)d_in[idx++];
  const float* mw1 = (const float*)d_in[idx++]; const float* mb1 = (const float*)d_in[idx++];
  const float* mw2 = (const float*)d_in[idx++]; const float* mb2 = (const float*)d_in[idx++];
  const float* lw1 = (const float*)d_in[idx++]; const float* lb1 = (const float*)d_in[idx++];
  const float* lw2 = (const float*)d_in[idx++]; const float* lb2 = (const float*)d_in[idx++];
  const float* fw = (const float*)d_in[idx++];
  const float* fb = (const float*)d_in[idx++];
  const float* rw1 = (const float*)d_in[idx++]; const float* rb1 = (const float*)d_in[idx++];
  const float* rw2 = (const float*)d_in[idx++]; const float* rb2 = (const float*)d_in[idx++];
  const float* log_decay = (const float*)d_in[idx++];
  const float* hww = (const float*)d_in[idx++];
  const float* hwb = (const float*)d_in[idx++];
  const float* hwr = (const float*)d_in[idx++];

  // ---- workspace layout (bytes) ----
  char* ws = (char*)d_ws;
  float*  h32    = (float*)(ws);                        // 33.5 MB
  float*  q32    = (float*)(ws + 33554432u);            // 33.5 MB
  float*  attn32 = (float*)(ws + 67108864u);            // 33.5 MB
  half_t* kT16   = (half_t*)(ws + 100663296u);          // 16 MB
  half_t* vT16   = (half_t*)(ws + 117440512u);          // 16 MB
  half_t* kvT16  = (half_t*)(ws + 134217728u);          // 2 MB
  half_t* WqT    = (half_t*)(ws + 136314880u);
  half_t* WkT    = WqT + 65536;
  half_t* WvT    = WkT + 65536;
  half_t* WoT    = WvT + 65536;
  float*  ksum   = (float*)(ws + 136839168u);
  float*  zbuf   = (float*)(ws + 136855552u);
  float*  s32    = q32;      // reuse after q consumed by attn_kernel
  float*  h3     = attn32;   // reuse after attn consumed by attnwo_kernel
  float*  out    = (float*)d_out;

  trans_kernel<<<256, 256, 0, stream>>>(Wq, WqT);
  trans_kernel<<<256, 256, 0, stream>>>(Wk, WkT);
  trans_kernel<<<256, 256, 0, stream>>>(Wv, WvT);
  trans_kernel<<<256, 256, 0, stream>>>(Wo, WoT);

  ConvParams CP;
  CP.x = x;
  CP.fc_dww = fc[0]; CP.fc_dwb = fc[1]; CP.fc_bn1g = fc[2]; CP.fc_bn1b = fc[3];
  CP.fc_pww = fc[4]; CP.fc_pwb = fc[5]; CP.fc_bn2g = fc[6]; CP.fc_bn2b = fc[7];
  for (int s = 0; s < 3; ++s) {
    CP.sc_dww[s] = sc[s][0]; CP.sc_dwb[s] = sc[s][1];
    CP.sc_bn1g[s] = sc[s][2]; CP.sc_bn1b[s] = sc[s][3];
    CP.sc_pww[s] = sc[s][4]; CP.sc_pwb[s] = sc[s][5];
    CP.sc_bn2g[s] = sc[s][6]; CP.sc_bn2b[s] = sc[s][7];
  }
  CP.scale_w = scale_w;
  CP.plw = plw; CP.plb = plb; CP.phw = phw; CP.phb = phb;
  CP.lng = lng; CP.lnb = lnb;
  CP.h32 = h32;
  conv_proj_kernel<<<32768, 128, 0, stream>>>(CP);

  qkv_kernel<<<256, 256, 0, stream>>>(h32, WqT, WkT, WvT, q32, kT16, vT16);
  ksum_kernel<<<4096, 256, 0, stream>>>(kT16, ksum);
  z_kernel<<<4096, 256, 0, stream>>>(q32, ksum, zbuf);
  kv_kernel<<<32, 256, 0, stream>>>(kT16, vT16, kvT16);
  attn_kernel<<<256, 256, 0, stream>>>(q32, kvT16, zbuf, attn32);
  attnwo_kernel<<<256, 256, 0, stream>>>(attn32, WoT, h32, s32);
  ln_kernel<<<4096, 256, 0, stream>>>(s32, aln_g, aln_b, h3);

  HeadParams HP;
  HP.h3 = h3; HP.x = x;
  HP.sw1 = sw1; HP.sb1 = sb1; HP.sw2 = sw2; HP.sb2 = sb2;
  HP.mw1 = mw1; HP.mb1 = mb1; HP.mw2 = mw2; HP.mb2 = mb2;
  HP.lw1 = lw1; HP.lb1 = lb1; HP.lw2 = lw2; HP.lb2 = lb2;
  HP.rw1 = rw1; HP.rb1 = rb1; HP.rw2 = rw2; HP.rb2 = rb2;
  HP.fw = fw; HP.fb = fb; HP.log_decay = log_decay;
  HP.hww = hww; HP.hwb = hwb; HP.hwr = hwr;
  HP.out = out;
  heads_kernel<<<4096, 256, 0, stream>>>(HP);
}
